// PluckerBigramAttention_60739427500733
// MI455X (gfx1250) — matrix operand order repaired and checked by reading, not compiled, not run on hardware
//
#include <hip/hip_runtime.h>

#define TT 2048
#define BB 4
#define DD 1024
#define HH 16
#define ROWS (BB * TT) /* 8192 */

typedef _Float16 h16;
typedef __attribute__((ext_vector_type(16))) _Float16 v16h;
typedef __attribute__((ext_vector_type(8)))  float    v8f;
typedef __attribute__((ext_vector_type(4)))  int      v4i;

// ---------------------------------------------------------------------------
// CDNA5 async global->LDS copies (ASYNCcnt path), probe-guarded.
// Builtin signature (from hipcc diagnostic): (as1 v4i*, as3 v4i*, imm, imm).
// ---------------------------------------------------------------------------
#if defined(__has_builtin)
#if __has_builtin(__builtin_amdgcn_global_load_async_to_lds_b128) && \
    __has_builtin(__builtin_amdgcn_s_wait_asynccnt)
#define HAVE_ASYNC_LDS 1
#endif
#endif
#ifndef HAVE_ASYNC_LDS
#define HAVE_ASYNC_LDS 0
#endif

#if HAVE_ASYNC_LDS
__device__ __forceinline__ void async_cp16(void* lds, const void* gsrc) {
  __builtin_amdgcn_global_load_async_to_lds_b128(
      (__attribute__((address_space(1))) v4i*)(void*)gsrc,
      (__attribute__((address_space(3))) v4i*)lds, 0, 0);
}
__device__ __forceinline__ void async_wait0() {
  __builtin_amdgcn_s_wait_asynccnt(0);
}
#else
__device__ __forceinline__ void async_cp16(void* lds, const void* gsrc) {
  *(uint4*)lds = *(const uint4*)gsrc;
}
__device__ __forceinline__ void async_wait0() {}
#endif

// ---------------------------------------------------------------------------
// WMMA helpers (CDNA5 wave32, V_WMMA_F32_16X16X32_F16)
// ---------------------------------------------------------------------------
__device__ __forceinline__ v8f wmma16(v16h a, v16h b, v8f c) {
  return __builtin_amdgcn_wmma_f32_16x16x32_f16(false, a, false, b, (short)0, c,
                                                false, false);
}

// A-fragment (16x32 f16) from row-major LDS tile; stride (in halves) must keep
// each 8-half group 16B-aligned so the pair lowers to two ds_load_b128.
__device__ __forceinline__ v16h ldsA_frag16(const h16* base, int stride) {
  const int l  = threadIdx.x & 31;
  const int m  = l & 15;
  const int kb = (l & 16) ? 16 : 0;
  const h16* p = base + m * stride + kb;
  union { uint4 u[2]; v16h v; } t;
  t.u[0] = *(const uint4*)(p);       // K = kb .. kb+7
  t.u[1] = *(const uint4*)(p + 8);   // K = kb+8 .. kb+15
  return t.v;
}

// B-fragment (32x16 f16, K-major rows) gathered from LDS (column reads).
__device__ __forceinline__ v16h ldsB_frag(const h16* base, int stride) {
  const int l  = threadIdx.x & 31;
  const int n  = l & 15;
  const int kb = (l & 16) ? 16 : 0;
  v16h b;
#pragma unroll
  for (int i = 0; i < 16; ++i) b[i] = base[(kb + i) * stride + n];
  return b;
}

// 16 contiguous halves from global (two b128 loads).
__device__ __forceinline__ v16h ldg16(const h16* p) {
  union { uint4 u[2]; v16h v; } t;
  t.u[0] = *(const uint4*)(p);
  t.u[1] = *(const uint4*)(p + 8);
  return t.v;
}

// ---------------------------------------------------------------------------
// Conversion kernels
// ---------------------------------------------------------------------------
__global__ void __launch_bounds__(256) cvt16x4_kernel(
    const float* __restrict__ src, h16* __restrict__ dst, int n4) {
  const int i = blockIdx.x * 256 + threadIdx.x;
  if (i < n4) {
    const float4 f = ((const float4*)src)[i];
    union { unsigned long long u; h16 h[4]; } t;
    t.h[0] = (h16)f.x; t.h[1] = (h16)f.y; t.h[2] = (h16)f.z; t.h[3] = (h16)f.w;
    ((unsigned long long*)dst)[i] = t.u;
  }
}

// (K,64) f32 -> (64,K) f16 transpose for projection weights.
__global__ void __launch_bounds__(256) wtrans_kernel(
    const float* __restrict__ src, h16* __restrict__ dst, int K) {
  const int i = blockIdx.x * 256 + threadIdx.x;
  if (i < K * 64) {
    const int k = i >> 6;
    const int c = i & 63;
    dst[(size_t)c * K + k] = (h16)src[i];
  }
}

// ---------------------------------------------------------------------------
// Kernel 1: fused q1/q2/k1/k2 projections + Plucker lines
// ---------------------------------------------------------------------------
__device__ __forceinline__ void exterior4(const float* p, const float* q,
                                          float* L) {
  L[0] = p[0] * q[1] - p[1] * q[0];
  L[1] = p[0] * q[2] - p[2] * q[0];
  L[2] = p[0] * q[3] - p[3] * q[0];
  L[3] = p[1] * q[2] - p[2] * q[1];
  L[4] = p[1] * q[3] - p[3] * q[1];
  L[5] = p[2] * q[3] - p[3] * q[2];
  float d = 0.0f;
#pragma unroll
  for (int i = 0; i < 6; ++i) d += L[i] * L[i];
  const float inv = 1.0f / fmaxf(sqrtf(d), 1e-12f);
#pragma unroll
  for (int i = 0; i < 6; ++i) L[i] *= inv;
}

__global__ void __launch_bounds__(256) proj_qk_kernel(
    const h16* __restrict__ xh, const h16* __restrict__ W1qt,
    const h16* __restrict__ W2qt, const h16* __restrict__ W1kt,
    const h16* __restrict__ W2kt, h16* __restrict__ Lq,
    h16* __restrict__ JLk) {
  __shared__ __align__(16) h16 sXa[16][40];    // x_t tile (16 x 32)
  __shared__ __align__(16) h16 sXb[16][40];    // x_{t-1} tile
  __shared__ __align__(16) float sP[16][260];  // q1|q2|k1|k2 (16 x 256)

  const int tid  = threadIdx.x;
  const int lane = tid & 31;
  const int wave = tid >> 5;
  const int g0   = blockIdx.x * 16;  // global row (b*T + t), T%16==0
  const int t0   = g0 & (TT - 1);
  const int arr  = wave >> 1;        // 0:W1q 1:W2q 2:W1k 3:W2k
  const int cb   = (wave & 1) * 32;
  const h16* Wp =
      (arr == 0) ? W1qt : (arr == 1) ? W2qt : (arr == 2) ? W1kt : W2kt;
  const bool isq = (arr < 2);
  const int  KW  = isq ? 2048 : 1024;  // transposed row length

  v8f acc0 = {0, 0, 0, 0, 0, 0, 0, 0};
  v8f acc1 = {0, 0, 0, 0, 0, 0, 0, 0};

  const int srow = (tid & 63) >> 2;  // staging row 0..15
  const int sseg = tid & 3;          // staging 16B segment 0..3

  for (int kk = 0; kk < 32; ++kk) {
    __syncthreads();
    // Stage x_t / x_{t-1} tiles: 2 x 16 rows x 32 halves = 128 x 16B.
    if (tid < 64) {
      async_cp16(&sXa[srow][sseg * 8],
                 xh + (size_t)(g0 + srow) * DD + kk * 32 + sseg * 8);
    } else if (tid < 128) {
      if (t0 + srow > 0) {
        async_cp16(&sXb[srow][sseg * 8],
                   xh + (size_t)(g0 + srow - 1) * DD + kk * 32 + sseg * 8);
      } else {  // first token of the batch: x_prev = 0 (one row, one block)
        *(uint4*)&sXb[srow][sseg * 8] = make_uint4(0, 0, 0, 0);
      }
    }
    async_wait0();
    __syncthreads();

    const v16h a0 = ldsA_frag16(&sXa[0][0], 40);
    v16h a1;
    if (isq) a1 = ldsA_frag16(&sXb[0][0], 40);

    const int n  = lane & 15;
    const int kb = (lane & 16) ? 16 : 0;
#pragma unroll
    for (int tb = 0; tb < 2; ++tb) {
      const int col = cb + tb * 16 + n;
      const h16* w0 = Wp + (size_t)col * KW + kk * 32 + kb;
      const v16h b0 = ldg16(w0);
      v8f& acc = tb ? acc1 : acc0;
      acc = wmma16(a0, b0, acc);
      if (isq) {
        const v16h b1 = ldg16(w0 + 1024);  // bigram second half of K
        acc = wmma16(a1, b1, acc);
      }
    }
  }

  // spill projections to LDS (C-frag: row = r + 8*(lane>=16), col = lane&15)
  {
    const int n  = lane & 15;
    const int rb = (lane & 16) ? 8 : 0;
#pragma unroll
    for (int tb = 0; tb < 2; ++tb) {
      const v8f& acc = tb ? acc1 : acc0;
      const int  col = arr * 64 + cb + tb * 16 + n;
#pragma unroll
      for (int r = 0; r < 8; ++r) sP[rb + r][col] = acc[r];
    }
  }
  __syncthreads();

  // Plucker stage: thread -> (row, head); float4 reads (16B aligned).
  const int r = tid >> 4;
  const int h = tid & 15;
  const float4 q1 = *(const float4*)&sP[r][h * 4];
  const float4 q2 = *(const float4*)&sP[r][64 + h * 4];
  const float4 k1 = *(const float4*)&sP[r][128 + h * 4];
  const float4 k2 = *(const float4*)&sP[r][192 + h * 4];
  const float q1a[4] = {q1.x, q1.y, q1.z, q1.w};
  const float q2a[4] = {q2.x, q2.y, q2.z, q2.w};
  const float k1a[4] = {k1.x, k1.y, k1.z, k1.w};
  const float k2a[4] = {k2.x, k2.y, k2.z, k2.w};
  float Lq6[6], Lk6[6];
  exterior4(q1a, q2a, Lq6);
  exterior4(k1a, k2a, Lk6);

  union { uint4 u; h16 hh[8]; } oq, ok;
#pragma unroll
  for (int d = 0; d < 6; ++d) oq.hh[d] = (h16)Lq6[d];
  oq.hh[6] = (h16)0; oq.hh[7] = (h16)0;
  // JL_k = (L5, -L4, L3, L2, -L1, L0)
  ok.hh[0] = (h16)Lk6[5];  ok.hh[1] = (h16)(-Lk6[4]);
  ok.hh[2] = (h16)Lk6[3];  ok.hh[3] = (h16)Lk6[2];
  ok.hh[4] = (h16)(-Lk6[1]); ok.hh[5] = (h16)Lk6[0];
  ok.hh[6] = (h16)0; ok.hh[7] = (h16)0;

  const size_t base = ((size_t)(g0 + r) * HH + h) * 8;
  *(uint4*)(Lq + base)  = oq.u;   // one b128 store each
  *(uint4*)(JLk + base) = ok.u;
}

// ---------------------------------------------------------------------------
// Kernel 2/4: GEMM  out = A(f16, MxK) @ W(f16, KxN) + bias
//   block 256 = 8 waves; tile 64x128; each wave 16x64 (4 WMMA / K-step)
// ---------------------------------------------------------------------------
__global__ void __launch_bounds__(256) gemm_kernel(
    const h16* __restrict__ A, const h16* __restrict__ W,
    const float* __restrict__ bias, float* __restrict__ outF,
    h16* __restrict__ outH, int M, int N, int K) {
  __shared__ __align__(16) h16 sA[64][40];    // 80B rows (16B aligned)
  __shared__ __align__(16) h16 sB[32][136];   // 272B rows (16B aligned)
  const int tid  = threadIdx.x;
  const int lane = tid & 31;
  const int wave = tid >> 5;
  const int n0   = blockIdx.x * 128;
  const int m0   = blockIdx.y * 64;
  const int wm   = wave >> 1;  // 0..3 -> 16 rows each
  const int wn   = wave & 1;   // 0..1 -> 64 cols each

  v8f acc[4];
#pragma unroll
  for (int t = 0; t < 4; ++t) acc[t] = (v8f){0, 0, 0, 0, 0, 0, 0, 0};

  for (int k0 = 0; k0 < K; k0 += 32) {
    __syncthreads();
    {  // A tile: 64 x 32 halves = 256 x 16B, one copy per thread
      const int row = tid >> 2, seg = tid & 3;
      async_cp16(&sA[row][seg * 8],
                 A + (size_t)(m0 + row) * K + k0 + seg * 8);
    }
#pragma unroll
    for (int e = 0; e < 2; ++e) {  // B tile: 32 x 128 halves = 512 x 16B
      const int j = tid + e * 256;
      const int row = j >> 4, seg = j & 15;
      async_cp16(&sB[row][seg * 8],
                 W + (size_t)(k0 + row) * N + n0 + seg * 8);
    }
    async_wait0();
    __syncthreads();

    const v16h a = ldsA_frag16(&sA[wm * 16][0], 40);
#pragma unroll
    for (int t = 0; t < 4; ++t) {
      const v16h b = ldsB_frag(&sB[0][wn * 64 + t * 16], 136);
      acc[t] = wmma16(a, b, acc[t]);
    }
  }

  const int rb = (lane & 16) ? 8 : 0;
#pragma unroll
  for (int t = 0; t < 4; ++t) {
    const int   n  = n0 + wn * 64 + t * 16 + (lane & 15);
    const float bv = bias[n];
#pragma unroll
    for (int r = 0; r < 8; ++r) {
      const size_t idx = (size_t)(m0 + wm * 16 + rb + r) * N + n;
      const float  v   = acc[t][r] + bv;
      if (outF) outF[idx] = v;
      else      outH[idx] = (h16)v;
    }
  }
}

// ---------------------------------------------------------------------------
// Kernel 3: causal flash attention on Plucker scores
//   block 128 = 4 waves, 64 queries/block (16 per wave), one (b,h) per block
// ---------------------------------------------------------------------------
__global__ void __launch_bounds__(128) attn_kernel(
    const h16* __restrict__ LqG, const h16* __restrict__ JLkG,
    const h16* __restrict__ Vh, h16* __restrict__ Oh) {
  __shared__ __align__(16) h16 sK[32][8];        // 32 keys x 8 line dims
  __shared__ __align__(16) h16 sV[32][72];       // 32 keys x 64 d_head
  __shared__ __align__(16) h16 sPP[4][16][40];   // per-wave P tile

  const int tid  = threadIdx.x;
  const int lane = tid & 31;
  const int wave = tid >> 5;
  const int qb   = blockIdx.x & 31;  // T/64 = 32 query blocks
  const int bh   = blockIdx.x >> 5;
  const int b    = bh >> 4;
  const int h    = bh & 15;
  const int Q0   = qb * 64;
  const int q0   = Q0 + wave * 16;
  const size_t rowbase = (size_t)b * TT;

  // Lq A-fragment: dims 0..7 in lanes 0-15 / elems 0-7; everything else 0.
  v16h aq;
#pragma unroll
  for (int i = 0; i < 16; ++i) aq[i] = (h16)0;
  if (lane < 16) {
    union { uint4 u; h16 hh[8]; } t;
    t.u = *(const uint4*)(LqG + ((rowbase + q0 + (lane & 15)) * HH + h) * 8);
#pragma unroll
    for (int i = 0; i < 8; ++i) aq[i] = t.hh[i];
  }

  float mrow[8], lrow[8];
  v8f   o[4];
  const v8f vzero = {0, 0, 0, 0, 0, 0, 0, 0};
#pragma unroll
  for (int r = 0; r < 8; ++r) { mrow[r] = -3.0e38f; lrow[r] = 0.0f; }
#pragma unroll
  for (int d = 0; d < 4; ++d) o[d] = vzero;

  const float SCALE = 0.4082482904638631f;  // 6^-0.5
  const int   KT    = Q0 / 32 + 2;
  const int   n     = lane & 15;
  const int   rb    = (lane & 16) ? 8 : 0;
  const int   kbB   = (lane & 16) ? 16 : 0;

  for (int kt = 0; kt < KT; ++kt) {
    __syncthreads();
    const int kbase = kt * 32;
    if (tid < 32)  // K lines: 32 x 16B
      async_cp16(&sK[tid][0],
                 JLkG + ((rowbase + kbase + tid) * HH + h) * 8);
#pragma unroll
    for (int e = 0; e < 2; ++e) {  // V tile: 32 x 64 halves = 256 x 16B
      const int j = tid + e * 128;
      async_cp16(&sV[j >> 3][(j & 7) * 8],
                 Vh + (rowbase + kbase + (j >> 3)) * DD + h * 64 + (j & 7) * 8);
    }
    async_wait0();
    __syncthreads();

    const bool active = (kbase <= q0 + 15);  // wave-uniform
    if (active) {
      v16h bk0, bk1;
#pragma unroll
      for (int i = 0; i < 16; ++i) {
        const int kd = kbB + i;
        h16 z0 = (h16)0, z1 = (h16)0;
        if (kd < 8) { z0 = sK[n][kd]; z1 = sK[16 + n][kd]; }
        bk0[i] = z0;
        bk1[i] = z1;
      }
      v8f s0 = wmma16(aq, bk0, vzero);
      v8f s1 = wmma16(aq, bk1, vzero);

#pragma unroll
      for (int r = 0; r < 8; ++r) {
        const int qrow = q0 + rb + r;
        float a = s0[r] * SCALE;
        float c = s1[r] * SCALE;
        if (kbase + n > qrow)      a = -3.0e38f;
        if (kbase + 16 + n > qrow) c = -3.0e38f;
        float mx = fmaxf(a, c);
        mx = fmaxf(mx, __shfl_xor(mx, 1, 16));
        mx = fmaxf(mx, __shfl_xor(mx, 2, 16));
        mx = fmaxf(mx, __shfl_xor(mx, 4, 16));
        mx = fmaxf(mx, __shfl_xor(mx, 8, 16));
        const float mnew  = fmaxf(mrow[r], mx);
        const float alpha = __expf(mrow[r] - mnew);
        mrow[r] = mnew;
        const float p0 = __expf(a - mnew);
        const float p1 = __expf(c - mnew);
        float rs = p0 + p1;
        rs += __shfl_xor(rs, 1, 16);
        rs += __shfl_xor(rs, 2, 16);
        rs += __shfl_xor(rs, 4, 16);
        rs += __shfl_xor(rs, 8, 16);
        lrow[r] = lrow[r] * alpha + rs;
#pragma unroll
        for (int d = 0; d < 4; ++d) o[d][r] *= alpha;
        sPP[wave][rb + r][n]      = (h16)p0;
        sPP[wave][rb + r][n + 16] = (h16)p1;
      }
    }
    __syncthreads();
    if (active) {
      const v16h pa = ldsA_frag16(&sPP[wave][0][0], 40);
#pragma unroll
      for (int d = 0; d < 4; ++d) {
        const v16h bv = ldsB_frag(&sV[0][d * 16], 72);
        o[d] = wmma16(pa, bv, o[d]);
      }
    }
  }

#pragma unroll
  for (int r = 0; r < 8; ++r) {
    const float  inv = 1.0f / lrow[r];
    const size_t row = rowbase + q0 + rb + r;
#pragma unroll
    for (int d = 0; d < 4; ++d)
      Oh[row * DD + h * 64 + d * 16 + n] = (h16)(o[d][r] * inv);
  }
}

// ---------------------------------------------------------------------------
// Host launch
// ---------------------------------------------------------------------------
extern "C" void kernel_launch(void* const* d_in, const int* in_sizes, int n_in,
                              void* d_out, int out_size, void* d_ws,
                              size_t ws_size, hipStream_t stream) {
  (void)in_sizes; (void)n_in; (void)out_size; (void)ws_size;
  const float* x   = (const float*)d_in[0];
  const float* W1q = (const float*)d_in[1];
  const float* W2q = (const float*)d_in[2];
  const float* W1k = (const float*)d_in[3];
  const float* W2k = (const float*)d_in[4];
  const float* Wv  = (const float*)d_in[5];
  const float* bv  = (const float*)d_in[6];
  const float* Wo  = (const float*)d_in[7];
  const float* bo  = (const float*)d_in[8];
  float* out = (float*)d_out;

  char* w = (char*)d_ws;
  const size_t SZ_X = (size_t)ROWS * DD * sizeof(h16);      // 16 MB
  const size_t SZ_L = (size_t)ROWS * HH * 8 * sizeof(h16);  // 2 MB
  const size_t SZ_WQ = (size_t)2048 * 64 * sizeof(h16);     // 256 KB
  const size_t SZ_WK = (size_t)1024 * 64 * sizeof(h16);     // 128 KB
  const size_t SZ_WD = (size_t)DD * DD * sizeof(h16);       // 2 MB
  size_t off = 0;
  h16* xh   = (h16*)(w + off); off += SZ_X;
  h16* Lq   = (h16*)(w + off); off += SZ_L;
  h16* JLk  = (h16*)(w + off); off += SZ_L;
  h16* vh   = (h16*)(w + off); off += SZ_X;
  h16* aoh  = (h16*)(w + off); off += SZ_X;
  h16* W1qt = (h16*)(w + off); off += SZ_WQ;
  h16* W2qt = (h16*)(w + off); off += SZ_WQ;
  h16* W1kt = (h16*)(w + off); off += SZ_WK;
  h16* W2kt = (h16*)(w + off); off += SZ_WK;
  h16* Wvh  = (h16*)(w + off); off += SZ_WD;
  h16* Woh  = (h16*)(w + off); off += SZ_WD;

  // Stage 0: convert / transpose everything to f16 once (L2-resident after).
  hipLaunchKernelGGL(cvt16x4_kernel, dim3(ROWS * DD / 4 / 256), dim3(256), 0,
                     stream, x, xh, ROWS * DD / 4);
  hipLaunchKernelGGL(cvt16x4_kernel, dim3(DD * DD / 4 / 256), dim3(256), 0,
                     stream, Wv, Wvh, DD * DD / 4);
  hipLaunchKernelGGL(cvt16x4_kernel, dim3(DD * DD / 4 / 256), dim3(256), 0,
                     stream, Wo, Woh, DD * DD / 4);
  hipLaunchKernelGGL(wtrans_kernel, dim3(2048 * 64 / 256), dim3(256), 0,
                     stream, W1q, W1qt, 2048);
  hipLaunchKernelGGL(wtrans_kernel, dim3(2048 * 64 / 256), dim3(256), 0,
                     stream, W2q, W2qt, 2048);
  hipLaunchKernelGGL(wtrans_kernel, dim3(1024 * 64 / 256), dim3(256), 0,
                     stream, W1k, W1kt, 1024);
  hipLaunchKernelGGL(wtrans_kernel, dim3(1024 * 64 / 256), dim3(256), 0,
                     stream, W2k, W2kt, 1024);

  // Stage 1: Plucker lines.
  hipLaunchKernelGGL(proj_qk_kernel, dim3(ROWS / 16), dim3(256), 0, stream, xh,
                     W1qt, W2qt, W1kt, W2kt, Lq, JLk);
  // Stage 2: V projection.
  hipLaunchKernelGGL(gemm_kernel, dim3(DD / 128, ROWS / 64), dim3(256), 0,
                     stream, xh, Wvh, bv, (float*)nullptr, vh, ROWS, DD, DD);
  // Stage 3: causal flash attention.
  hipLaunchKernelGGL(attn_kernel, dim3(BB * HH * (TT / 64)), dim3(128), 0,
                     stream, Lq, JLk, vh, aoh);
  // Stage 4: output projection (f32 out).
  hipLaunchKernelGGL(gemm_kernel, dim3(DD / 128, ROWS / 64), dim3(256), 0,
                     stream, aoh, Woh, bo, out, (h16*)nullptr, ROWS, DD, DD);
}
